// FourierLayer2dLite_7009386627310
// MI455X (gfx1250) — matrix operand order repaired and checked by reading, not compiled, not run on hardware
//
#include <hip/hip_runtime.h>
#include <math.h>

typedef __attribute__((ext_vector_type(16))) __bf16 v16bf;
typedef __attribute__((ext_vector_type(8)))  __bf16 v8bf;
typedef __attribute__((ext_vector_type(8)))  float  v8f;
typedef __attribute__((ext_vector_type(4)))  unsigned int u32x4;
typedef __attribute__((ext_vector_type(8)))  int i32x8;
typedef __attribute__((ext_vector_type(4)))  int i32x4;

// Problem dims
#define Bn 16
#define Mn 256
#define Nn 256
#define In 64
#define On 64

// Workspace layout (bytes). All offsets 16B-aligned.
static constexpr size_t OFF_F1 = 0;                                // F1  [32][256] bf16   (fwd n-DFT, scaled 1/256)
static constexpr size_t OFF_A2 = OFF_F1 + (size_t)32*256*2;        // A2  [64][512]        (fwd m-DFT, complex-embedded)
static constexpr size_t OFF_A4 = OFF_A2 + (size_t)64*512*2;        // A4  [512][64]        (inv m-DFT, complex-embedded)
static constexpr size_t OFF_A5 = OFF_A4 + (size_t)512*64*2;        // A5  [256][32]        (inv rfft over n, scaled)
static constexpr size_t OFF_WB = OFF_A5 + (size_t)256*32*2;        // WBt [o64][i64]       (W transposed)
static constexpr size_t OFF_G  = OFF_WB + (size_t)64*64*2;         // Gt  [512][n128][k128] (channel mix, transposed)
static constexpr size_t OFF_T1 = OFF_G  + (size_t)512*128*128*2;   // T1  [b16][kn16][i64][c2][m256]  (B^T for stage B)
static constexpr size_t OFF_C2 = OFF_T1 + (size_t)16*16*64*512*2;  // C2  [mode512][b16][2i+c 128]    (A for stage C)
static constexpr size_t OFF_C3 = OFF_C2 + (size_t)512*16*128*2;    // C3  [b16][kn16][o64][2km+c 64]  (B^T for stage D)
static constexpr size_t OFF_Z  = OFF_C3 + (size_t)16*16*64*64*2;   // Z   [b16][m256][o64][2kn+c 32]  (B^T for stage EF)

static __device__ __forceinline__ __bf16 f2bf(float f) {
  unsigned u = __builtin_bit_cast(unsigned, f);
  u += 0x7FFFu + ((u >> 16) & 1u);  // round-to-nearest-even
  unsigned short h = (unsigned short)(u >> 16);
  return __builtin_bit_cast(__bf16, h);
}

static __device__ __forceinline__ v16bf catbf(v8bf lo, v8bf hi) {
  return __builtin_shufflevector(lo, hi, 0, 1, 2, 3, 4, 5, 6, 7, 8, 9, 10, 11, 12, 13, 14, 15);
}

// LDS byte offset of a __shared__ object (generic -> AS(3) addrspacecast).
static __device__ __forceinline__ unsigned lds_addr32(const void* p) {
  const __attribute__((address_space(3))) void* lp =
      (const __attribute__((address_space(3))) void*)p;
  return (unsigned)(unsigned long long)lp;
}

// Tensor Data Mover: 1-D bf16 tile, global -> LDS. D# per CDNA5 ISA 8.3/8.4:
// group0: count=1, lds_addr, global_addr[56:0], type=2. group1: data_size=2B,
// tensor_dim0 = tile_dim0 = nelem, tensor_dim0_stride = nelem, no pad/iterate/multicast.
// Issue from ONE wave only (TDM ignores EXEC); completion via TENSORcnt.
static __device__ __forceinline__ void tdm_load_bf16(__bf16* lds, const __bf16* gsrc,
                                                     unsigned nelem /* <= 65535 */) {
  unsigned long long ga = (unsigned long long)gsrc;
  u32x4 g0 = {1u,                                   // count=1, user descriptor
              lds_addr32(lds),                      // lds_addr (bytes)
              (unsigned)(ga & 0xFFFFFFFFu),         // global_addr[31:0]
              (unsigned)((ga >> 32) & 0x01FFFFFFu) | (2u << 30)};  // addr[56:32] | type=2
  i32x8 g1 = {(int)(1u << 16),                      // wg_mask=0, data_size=1 (2 bytes)
              (int)((nelem & 0xFFFFu) << 16),       // tensor_dim0[15:0] in bits 63:48
              (int)((nelem >> 16) & 0xFFFFu),       // tensor_dim0[31:16]; tensor_dim1=0
              (int)((nelem & 0xFFFFu) << 16),       // tile_dim0 in bits 127:112
              0,                                    // tile_dim1=0, tile_dim2=0 (1-D)
              (int)nelem,                           // tensor_dim0_stride[31:0]
              0, 0};
  i32x4 z4 = {0, 0, 0, 0};
  i32x8 z8 = {0, 0, 0, 0, 0, 0, 0, 0};
  __builtin_amdgcn_tensor_load_to_lds(g0, g1, z4, z4, z8, 0);  // 6-arg form
}

// One 16x16 tile, contracting K (mult of 32).
// A: row-major [rows x lda] (lda % 8 == 0). Bt: TRANSPOSED, [cols x ldk] (ldk % 8 == 0).
// Each fragment = two 16B ds_load_b128 per operand. Layouts per CDNA5 ISA 7.12.2 (wave32).
static __device__ __forceinline__ void wmma_kpanel(const __bf16* As, int lda, int arow,
                                                   const __bf16* Bt, int ldk, int bcol,
                                                   int K, v8f& acc) {
  const int lane = threadIdx.x & 31;
  const int half = lane >> 4;
  const int r    = lane & 15;
  const __bf16* ap = As + (arow + r) * lda;
  const __bf16* bp = Bt + (bcol + r) * ldk;
  for (int k0 = 0; k0 < K; k0 += 32) {
    v8bf a0 = *(const v8bf*)(ap + k0 + half * 8);
    v8bf a1 = *(const v8bf*)(ap + k0 + 16 + half * 8);
    v8bf b0 = *(const v8bf*)(bp + k0 + half * 8);
    v8bf b1 = *(const v8bf*)(bp + k0 + 16 + half * 8);
    acc = __builtin_amdgcn_wmma_f32_16x16x32_bf16(false, catbf(a0, a1), false, catbf(b0, b1),
                                                  (short)0, acc, false, false);
  }
}

// Contiguous bf16 copy, 16 bytes per op (global_load_b128 -> ds_store_b128).
static __device__ __forceinline__ void copy_bf16(__bf16* dst, const __bf16* src, int nelem) {
  v8bf* d = (v8bf*)dst;
  const v8bf* s = (const v8bf*)src;
  for (int i = threadIdx.x; i < nelem / 8; i += 256) d[i] = s[i];
}

// ---------------------------------------------------------------- twiddles + W^T
__global__ __launch_bounds__(256) void k_init(const float* Wf, char* ws) {
  __bf16* F1 = (__bf16*)(ws + OFF_F1);
  __bf16* A2 = (__bf16*)(ws + OFF_A2);
  __bf16* A4 = (__bf16*)(ws + OFF_A4);
  __bf16* A5 = (__bf16*)(ws + OFF_A5);
  __bf16* WB = (__bf16*)(ws + OFF_WB);
  const float TWO_PI = 6.28318530717958647692f;
  const int tid = blockIdx.x * blockDim.x + threadIdx.x;
  const int nth = gridDim.x * blockDim.x;

  for (int idx = tid; idx < 32 * 256; idx += nth) {
    int f = idx >> 8, n = idx & 255;
    int kn = f >> 1, c = f & 1;
    float th = TWO_PI * (float)((kn * n) & 255) / 256.0f;
    float s = 1.0f / 256.0f;
    F1[idx] = f2bf(c == 0 ? cosf(th) * s : -sinf(th) * s);
  }
  for (int idx = tid; idx < 64 * 512; idx += nth) {
    int g = idx >> 9, col = idx & 511;
    int km = g >> 1, cp = g & 1;
    int c = col >> 8, m = col & 255;
    int f = (km < 16) ? km : 240 + (km - 16);
    float th = TWO_PI * (float)((f * m) & 255) / 256.0f;
    float cs = cosf(th), sn = sinf(th);
    float v = (cp == 0) ? (c == 0 ? cs : sn) : (c == 0 ? -sn : cs);
    A2[idx] = f2bf(v);
  }
  for (int idx = tid; idx < 512 * 64; idx += nth) {
    int row = idx >> 6, col = idx & 63;
    int c = row >> 8, m = row & 255;
    int km = col >> 1, cp = col & 1;
    int f = (km < 16) ? km : 240 + (km - 16);
    float th = TWO_PI * (float)((f * m) & 255) / 256.0f;
    float cs = cosf(th), sn = sinf(th);
    float v = (c == 0) ? (cp == 0 ? cs : -sn) : (cp == 0 ? sn : cs);
    A4[idx] = f2bf(v);
  }
  for (int idx = tid; idx < 256 * 32; idx += nth) {
    int n = idx >> 5, col = idx & 31;
    int kn = col >> 1, cp = col & 1;
    float th = TWO_PI * (float)((kn * n) & 255) / 256.0f;
    float w = (kn == 0 ? 1.0f : 2.0f) * sqrtf(256.0f * 129.0f) / 65536.0f;
    A5[idx] = f2bf(cp == 0 ? w * cosf(th) : -w * sinf(th));
  }
  for (int idx = tid; idx < 64 * 64; idx += nth) {
    int o = idx >> 6, i = idx & 63;
    WB[idx] = f2bf(Wf[i * 64 + o]);  // WBt[o][i]
  }
}

// -------------------------------- per-mode complex weight embedding, TRANSPOSED
// Gt[mode][n=2o+c'][k=2i+c]
__global__ __launch_bounds__(256) void k_gw(const float* fw1, const float* fw2, char* ws) {
  __bf16* G = (__bf16*)(ws + OFF_G);
  const size_t total = (size_t)512 * 128 * 128;
  const size_t tid = (size_t)blockIdx.x * blockDim.x + threadIdx.x;
  const size_t nth = (size_t)gridDim.x * blockDim.x;
  for (size_t idx = tid; idx < total; idx += nth) {
    int mode = (int)(idx >> 14);
    int n = (int)((idx >> 7) & 127);
    int k = (int)(idx & 127);
    int km = mode >> 4, kn = mode & 15;
    int i = k >> 1, c = k & 1;
    int o = n >> 1, cp = n & 1;
    const float* fw = (km < 16) ? fw1 : fw2;
    int xw = (km < 16) ? km : km - 16;
    size_t base = ((((size_t)i * 64 + o) * 16 + xw) * 16 + kn) * 2;
    float wr = fw[base], wi = fw[base + 1];
    float v = (cp == 0) ? (c == 0 ? wr : -wi) : (c == 0 ? wi : wr);
    G[idx] = f2bf(v);
  }
}

// ------------------------------------------------ stage A: fwd DFT over n
// per (b,m): [32x256] x [256x64] -> T1[b][kn][i][c][m]   (B^T for stage B)
__global__ __launch_bounds__(256) void k_stageA(const float* x, char* ws) {
  __shared__ alignas(16) __bf16 As[32 * 64];
  __shared__ alignas(16) __bf16 Bt[64 * 64];
  const __bf16* F1 = (const __bf16*)(ws + OFF_F1);
  __bf16* T1 = (__bf16*)(ws + OFF_T1);
  const int blk = blockIdx.x, b = blk >> 8, m = blk & 255;
  const float* xs = x + (((size_t)b * 256 + m) * 256) * 64;
  const int tid = threadIdx.x, w = tid >> 5, lane = tid & 31;
  const int half = lane >> 4, r = lane & 15;
  const int arow = (w >> 2) * 16, bcol = (w & 3) * 16;
  v8f acc = {};
  for (int kc = 0; kc < 256; kc += 64) {
    for (int idx = tid; idx < 256; idx += 256) {
      int row = idx >> 3, c8 = (idx & 7) * 8;
      *(v8bf*)&As[row * 64 + c8] = *(const v8bf*)&F1[row * 256 + kc + c8];
    }
    for (int idx = tid; idx < 1024; idx += 256) {
      int n = idx >> 4;
      int i0 = (idx & 15) * 4;
      const float4 xv = *(const float4*)(xs + (size_t)(kc + n) * 64 + i0);
      Bt[(i0 + 0) * 64 + n] = f2bf(xv.x);
      Bt[(i0 + 1) * 64 + n] = f2bf(xv.y);
      Bt[(i0 + 2) * 64 + n] = f2bf(xv.z);
      Bt[(i0 + 3) * 64 + n] = f2bf(xv.w);
    }
    if (kc + 64 < 256)  // prefetch next x chunk -> global_prefetch_b8
      __builtin_prefetch(xs + (size_t)(kc + 64) * 64 + tid * 16, 0, 1);
    __syncthreads();
    wmma_kpanel(As, 64, arow, Bt, 64, bcol, 64, acc);
    __syncthreads();
  }
#pragma unroll
  for (int v = 0; v < 8; ++v) {
    int f = arow + v + half * 8;
    int kn = f >> 1, c = f & 1;
    int i = bcol + r;
    size_t off = ((((size_t)b * 16 + kn) * 64 + i) * 2 + c) * 256 + m;
    T1[off] = f2bf(acc[v]);
  }
}

// ------------------------------------------------ stage B: fwd DFT over m
// per (b,kn): [64x512] x [512x64] -> C2[mode][b][2i+c]   (A for stage C)
__global__ __launch_bounds__(256) void k_stageB(char* ws) {
  __shared__ alignas(16) __bf16 As[64 * 64];
  __shared__ alignas(16) __bf16 Bt[64 * 64];
  const __bf16* A2 = (const __bf16*)(ws + OFF_A2);
  const __bf16* T1 = (const __bf16*)(ws + OFF_T1);
  __bf16* C2 = (__bf16*)(ws + OFF_C2);
  const int blk = blockIdx.x, b = blk >> 4, kn = blk & 15;
  const __bf16* slab = T1 + ((size_t)b * 16 + kn) * 64 * 512;  // [i64][k512], already B^T
  const int tid = threadIdx.x, w = tid >> 5, lane = tid & 31;
  const int half = lane >> 4, r = lane & 15;
  const int tr = (w & 3) * 16, tc0 = (w >> 2) * 32;
  v8f acc0 = {}, acc1 = {};
  for (int kc = 0; kc < 512; kc += 64) {
    for (int idx = tid; idx < 512; idx += 256) {
      int row = idx >> 3, c8 = (idx & 7) * 8;
      *(v8bf*)&As[row * 64 + c8] = *(const v8bf*)&A2[row * 512 + kc + c8];
      *(v8bf*)&Bt[row * 64 + c8] = *(const v8bf*)&slab[(size_t)row * 512 + kc + c8];
    }
    __syncthreads();
    wmma_kpanel(As, 64, tr, Bt, 64, tc0, 64, acc0);
    wmma_kpanel(As, 64, tr, Bt, 64, tc0 + 16, 64, acc1);
    __syncthreads();
  }
#pragma unroll
  for (int v = 0; v < 8; ++v) {
    int g = tr + v + half * 8;
    int km = g >> 1, cp = g & 1;
    size_t mode = (size_t)km * 16 + kn;
    C2[(mode * 16 + b) * 128 + 2 * (tc0 + r) + cp] = f2bf(acc0[v]);
    C2[(mode * 16 + b) * 128 + 2 * (tc0 + 16 + r) + cp] = f2bf(acc1[v]);
  }
}

// ------------------------------------------------ stage C: per-mode channel mix
// per mode: [16x128] x [128x128] -> C3[b][kn][o][2km+c']   (B^T for stage D)
__global__ __launch_bounds__(256) void k_stageC(char* ws) {
  __shared__ alignas(16) __bf16 As[16 * 128];
  __shared__ alignas(16) __bf16 Bt[128 * 128];
  const int mode = blockIdx.x;
  const __bf16* C2 = (const __bf16*)(ws + OFF_C2) + (size_t)mode * 16 * 128;
  const __bf16* G = (const __bf16*)(ws + OFF_G) + (size_t)mode * 128 * 128;
  __bf16* C3 = (__bf16*)(ws + OFF_C3);
  const int tid = threadIdx.x, w = tid >> 5, lane = tid & 31;
  const int half = lane >> 4, r = lane & 15;
  if (tid < 32) tdm_load_bf16(Bt, G, 128 * 128);   // TDM: 32KB weight panel
  copy_bf16(As, C2, 16 * 128);
  if (tid < 32) __builtin_amdgcn_s_wait_tensorcnt(0);
  __syncthreads();
  v8f acc = {};
  wmma_kpanel(As, 128, 0, Bt, 128, w * 16, 128, acc);
  const int km = mode >> 4, kn = mode & 15;
#pragma unroll
  for (int v = 0; v < 8; ++v) {
    int b = v + half * 8;
    int n = w * 16 + r;
    int o = n >> 1, cp = n & 1;
    C3[(((size_t)b * 16 + kn) * 64 + o) * 64 + 2 * km + cp] = f2bf(acc[v]);
  }
}

// ------------------------------------------------ stage D: inverse DFT over m
// per (b,kn): [512x64] x [64x64] -> Z[b][m][o][2kn+c]   (B^T for stage EF)
__global__ __launch_bounds__(256) void k_stageD(char* ws) {
  __shared__ alignas(16) __bf16 As[64 * 64];
  __shared__ alignas(16) __bf16 Bt[64 * 64];
  const __bf16* A4 = (const __bf16*)(ws + OFF_A4);
  const __bf16* C3 = (const __bf16*)(ws + OFF_C3);
  __bf16* Z = (__bf16*)(ws + OFF_Z);
  const int blk = blockIdx.x, b = blk >> 4, kn = blk & 15;
  const __bf16* slab = C3 + ((size_t)b * 16 + kn) * 64 * 64;  // [o64][k64], already B^T
  const int tid = threadIdx.x, w = tid >> 5, lane = tid & 31;
  const int half = lane >> 4, r = lane & 15;
  const int tr = (w & 3) * 16, tc0 = (w >> 2) * 32;
  copy_bf16(Bt, slab, 64 * 64);
  __syncthreads();
  for (int rt = 0; rt < 512; rt += 64) {
    if (tid < 32) {  // TDM: 8KB twiddle chunk
      tdm_load_bf16(As, A4 + (size_t)rt * 64, 64 * 64);
      __builtin_amdgcn_s_wait_tensorcnt(0);
    }
    __syncthreads();
    v8f acc0 = {}, acc1 = {};
    wmma_kpanel(As, 64, tr, Bt, 64, tc0, 64, acc0);
    wmma_kpanel(As, 64, tr, Bt, 64, tc0 + 16, 64, acc1);
#pragma unroll
    for (int v = 0; v < 8; ++v) {
      int row = rt + tr + v + half * 8;
      int c = row >> 8, m = row & 255;
      size_t zb = (((size_t)b * 256 + m) * 64);
      Z[(zb + tc0 + r) * 32 + 2 * kn + c] = f2bf(acc0[v]);
      Z[(zb + tc0 + 16 + r) * 32 + 2 * kn + c] = f2bf(acc1[v]);
    }
    __syncthreads();
  }
}

// ------------------------------------ stage E+F fused: inverse rfft over n, then
// relu(y@W + b + y) streamed to d_out. per (b,m): [256x32]x[32x64] then [256x64]x[64x64]
__global__ __launch_bounds__(256) void k_stageEF(const float* bias, float* out, char* ws) {
  __shared__ alignas(16) __bf16 A5s[256 * 32];
  __shared__ alignas(16) __bf16 Zt[64 * 32];
  __shared__ alignas(16) __bf16 Ybf[256 * 64];
  __shared__ alignas(16) __bf16 Wt[64 * 64];
  const __bf16* A5 = (const __bf16*)(ws + OFF_A5);
  const __bf16* Z = (const __bf16*)(ws + OFF_Z);
  const __bf16* WB = (const __bf16*)(ws + OFF_WB);
  const int blk = blockIdx.x, b = blk >> 8, m = blk & 255;
  const __bf16* zslab = Z + ((size_t)b * 256 + m) * 64 * 32;
  const int tid = threadIdx.x, w = tid >> 5, lane = tid & 31;
  const int half = lane >> 4, r = lane & 15;
  if (tid < 32) tdm_load_bf16(A5s, A5, 256 * 32);  // TDM: 16KB twiddle panel
  copy_bf16(Zt, zslab, 64 * 32);
  copy_bf16(Wt, WB, 64 * 64);
  if (tid < 32) __builtin_amdgcn_s_wait_tensorcnt(0);
  __syncthreads();

  v8f accE[8];
#pragma unroll
  for (int t = 0; t < 8; ++t) {
    int rt = (w * 2 + (t >> 2)) * 16;
    int tc = (t & 3) * 16;
    v8f a = {};
    wmma_kpanel(A5s, 32, rt, Zt, 32, tc, 32, a);
    accE[t] = a;
#pragma unroll
    for (int v = 0; v < 8; ++v) Ybf[(rt + v + half * 8) * 64 + tc + r] = f2bf(a[v]);
  }
  __syncthreads();
#pragma unroll
  for (int t = 0; t < 8; ++t) {
    int rt = (w * 2 + (t >> 2)) * 16;
    int tc = (t & 3) * 16;
    v8f a = {};
    wmma_kpanel(Ybf, 64, rt, Wt, 64, tc, 64, a);
#pragma unroll
    for (int v = 0; v < 8; ++v) {
      int n = rt + v + half * 8;
      int o = tc + r;
      float val = a[v] + accE[t][v] + bias[o];
      val = val > 0.0f ? val : 0.0f;
      out[(((size_t)b * 256 + m) * 256 + n) * 64 + o] = val;
    }
  }
}

extern "C" void kernel_launch(void* const* d_in, const int* in_sizes, int n_in,
                              void* d_out, int out_size, void* d_ws, size_t ws_size,
                              hipStream_t stream) {
  (void)in_sizes; (void)n_in; (void)out_size; (void)ws_size;
  const float* x    = (const float*)d_in[0];
  const float* fw1  = (const float*)d_in[1];
  const float* fw2  = (const float*)d_in[2];
  const float* Wf   = (const float*)d_in[3];
  const float* bias = (const float*)d_in[4];
  float* out = (float*)d_out;
  char* ws = (char*)d_ws;

  k_init<<<64, 256, 0, stream>>>(Wf, ws);
  k_gw<<<4096, 256, 0, stream>>>(fw1, fw2, ws);
  k_stageA<<<Bn * Mn, 256, 0, stream>>>(x, ws);
  k_stageB<<<Bn * 16, 256, 0, stream>>>(ws);
  k_stageC<<<512, 256, 0, stream>>>(ws);
  k_stageD<<<Bn * 16, 256, 0, stream>>>(ws);
  k_stageEF<<<Bn * Mn, 256, 0, stream>>>(bias, out, ws);
}
